// Non_Local_Module_11605001634104
// MI455X (gfx1250) — matrix operand order repaired and checked by reading, not compiled, not run on hardware
//
#include <hip/hip_runtime.h>

// Problem constants (match reference)
#define Bn 8
#define Cc 256
#define Nn 4096   // H*W
#define O2 512    // 2*C

typedef _Float16      v16h __attribute__((ext_vector_type(16)));
typedef float         v8f  __attribute__((ext_vector_type(8)));
typedef unsigned int  v4u  __attribute__((ext_vector_type(4)));
typedef int           v4i_ __attribute__((ext_vector_type(4)));
typedef float         v2f  __attribute__((ext_vector_type(2)));

union Frag16 { v16h v; v4u q[2]; };

// ---- gfx1250 async global->LDS path (guarded; falls back to plain copy) ----
#if defined(__has_builtin)
#if __has_builtin(__builtin_amdgcn_global_load_async_to_lds_b128) && \
    __has_builtin(__builtin_amdgcn_s_wait_asynccnt)
#define HAS_ASYNC 1
#endif
#endif

#ifdef HAS_ASYNC
#pragma message("gfx1250: using global_load_async_to_lds_b128 + s_wait_asynccnt")
#define ASYNC_CP(gp, lp)                                                   \
  __builtin_amdgcn_global_load_async_to_lds_b128(                          \
      (__attribute__((address_space(1))) v4i_*)(gp),                       \
      (__attribute__((address_space(3))) v4i_*)(lp), 0, 0)
#define ASYNC_WAIT(n) __builtin_amdgcn_s_wait_asynccnt(n)
#else
#pragma message("async-to-LDS builtins unavailable: synchronous staging fallback")
#define ASYNC_CP(gp, lp) (*(v4u*)(lp) = *(const v4u*)(gp))
#define ASYNC_WAIT(n) ((void)0)
#endif

static __device__ inline v8f wmma_f16(v16h a, v16h b, v8f c) {
  // D = A(16x32 f16) x B(32x16 f16) + C(16x16 f32)
  return __builtin_amdgcn_wmma_f32_16x16x32_f16(
      /*neg_a=*/false, a, /*neg_b=*/false, b,
      /*c_mod=*/(short)0, c, /*reuse_a=*/false, /*reuse_b=*/false);
}

// ---------------------------------------------------------------------------
// Projection: out = x^T * W^T + bias   (1x1 conv == channel GEMM), f32 -> f16
//   x: (B, C, N) f32,  W: (O, C) f32,  bias: (O,) f32
//   storeT == 0: out (B, N, O)  [Q, K^T layouts]
//   storeT == 1: out (B, O, N)  [V layout]
// ---------------------------------------------------------------------------
__global__ __launch_bounds__(256) void proj_kernel(
    const float* __restrict__ x,
    const float* __restrict__ W,
    const float* __restrict__ bias,
    _Float16* __restrict__ out,
    int O, int storeT)
{
  __shared__ __align__(16) _Float16 ldsW[128][256]; // [o][c], c contiguous
  __shared__ __align__(16) _Float16 ldsA[16][256];  // [n][c], c contiguous

  const int tid   = threadIdx.x;
  const int b     = blockIdx.z;
  const int nBase = blockIdx.x * 16;
  const int oBase = blockIdx.y * 128;

  // Stage W block (f32 -> f16): ldsW[o][c] = W[oBase+o][c]
  {
    const v2f* Wv = (const v2f*)(W + (size_t)oBase * Cc);
    #pragma unroll
    for (int k = 0; k < 64; ++k) {
      int idx = tid + k * 256;        // 0..16383 float2's
      int o   = idx >> 7;
      int cp  = idx & 127;
      v2f w2  = Wv[o * 128 + cp];
      ldsW[o][cp * 2]     = (_Float16)w2.x;
      ldsW[o][cp * 2 + 1] = (_Float16)w2.y;
    }
  }
  // Stage x^T tile: ldsA[n][c] = x[b][c][nBase+n]
  {
    const float* xb = x + (size_t)b * Cc * Nn;
    #pragma unroll
    for (int k = 0; k < 16; ++k) {
      int idx = tid + k * 256;        // 0..4095
      int c = idx >> 4;
      int n = idx & 15;
      ldsA[n][c] = (_Float16)xb[(size_t)c * Nn + nBase + n];
    }
  }
  __syncthreads();

  const int wave = tid >> 5;
  const int lane = tid & 31;
  const int col  = lane & 15;
  const int g    = lane >> 4;   // half-wave select
  const int oLoc = wave * 16;

  v8f acc = {};
  #pragma unroll
  for (int cs = 0; cs < 8; ++cs) {  // K-dim: C=256 in steps of 32
    Frag16 a, bm;
    a.q[0]  = *(const v4u*)&ldsA[col][cs * 32 + g * 8];
    a.q[1]  = *(const v4u*)&ldsA[col][cs * 32 + 16 + g * 8];
    bm.q[0] = *(const v4u*)&ldsW[oLoc + col][cs * 32 + g * 8];
    bm.q[1] = *(const v4u*)&ldsW[oLoc + col][cs * 32 + 16 + g * 8];
    acc = wmma_f16(a.v, bm.v, acc);
  }

  const int   oG = oBase + oLoc + col;
  const float bb = bias[oG];
  if (storeT == 0) {
    _Float16* op = out + ((size_t)b * Nn + nBase + g * 8) * O + oG;
    #pragma unroll
    for (int r = 0; r < 8; ++r)
      op[(size_t)r * O] = (_Float16)(acc[r] + bb);
  } else {
    union { v4u q; _Float16 h[8]; } st;
    #pragma unroll
    for (int r = 0; r < 8; ++r) st.h[r] = (_Float16)(acc[r] + bb);
    *(v4u*)(out + ((size_t)b * O + oG) * Nn + nBase + g * 8) = st.q;
  }
}

// ---------------------------------------------------------------------------
// Flash attention + residual, double-buffered async K/V staging.
//   Q  (B, N, 512) f16, Kt (B, N, 512) f16 (= K^T), V (B, 256, N) f16
//   out[b][c][n] = sum_j softmax_j(Q K)[n][j] * V[c][j]  + x[b][c][n]
// ---------------------------------------------------------------------------
__global__ __launch_bounds__(256) void attn_kernel(
    const _Float16* __restrict__ Q,
    const _Float16* __restrict__ Kt,
    const _Float16* __restrict__ V,
    const float* __restrict__ x,
    float* __restrict__ outp)
{
  __shared__ __align__(16) _Float16 ldsK[2][32][512];   // [buf][j][o] 64 KB
  __shared__ __align__(16) _Float16 ldsV[2][256][32];   // [buf][c][j] 32 KB
  __shared__ __align__(16) _Float16 ldsP[8][16][32];    // per-wave P scratch

  const int tid     = threadIdx.x;
  const int wave    = tid >> 5;
  const int lane    = tid & 31;
  const int col     = lane & 15;
  const int g       = lane >> 4;
  const int b       = blockIdx.y;
  const int rowBase = blockIdx.x * 128 + wave * 16;

  const _Float16* Kb = Kt + (size_t)b * Nn * O2;
  const _Float16* Vb = V  + (size_t)b * Cc * Nn;

  // Issue K/V tile copy (12 x b128 per thread) into buffer `buf`.
  auto stage = [&](int buf, int js) {
    const v4u* srcK = (const v4u*)(Kb + (size_t)js * O2);   // 2048 chunks
    v4u* dstK = (v4u*)&ldsK[buf][0][0];
    #pragma unroll
    for (int k = 0; k < 8; ++k) {
      int idx = tid + k * 256;
      ASYNC_CP(srcK + idx, dstK + idx);
    }
    #pragma unroll
    for (int k = 0; k < 4; ++k) {
      int idx  = tid + k * 256;        // 1024 chunks of 8 halves
      int c    = idx >> 2;
      int part = idx & 3;
      ASYNC_CP((const v4u*)(Vb + (size_t)c * Nn + js + part * 8),
               (v4u*)&ldsV[buf][c][part * 8]);
    }
  };

  // Resident Q fragments: 16 chunks of 32 along the o-dim (128 VGPRs)
  Frag16 qf[16];
  {
    const _Float16* qrow = Q + ((size_t)b * Nn + rowBase + col) * O2;
    #pragma unroll
    for (int os = 0; os < 16; ++os) {
      qf[os].q[0] = *(const v4u*)(qrow + os * 32 + g * 8);
      qf[os].q[1] = *(const v4u*)(qrow + os * 32 + 16 + g * 8);
    }
  }

  v8f Oacc[16] = {};           // O tile 16 x 256 (f32)
  float mrow[8], lrow[8];      // running max / sum for rows r + g*8
  #pragma unroll
  for (int r = 0; r < 8; ++r) { mrow[r] = -1e30f; lrow[r] = 0.f; }

  stage(0, 0);                 // prologue: tile 0 in flight
  int cur = 0;

  for (int js = 0; js < Nn; js += 32) {
    __syncthreads();           // everyone done reading buf[cur^1]
    if (js + 32 < Nn) {
      stage(cur ^ 1, js + 32); // overlap next tile with this tile's compute
      ASYNC_WAIT(12);          // cur's 12 chunks (issued earlier) complete
    } else {
      ASYNC_WAIT(0);
    }
    __syncthreads();           // all waves' cur copies visible

    // S = Q * K^T : two 16x16 f32 tiles (j columns js..js+31)
    v8f s[2];
    #pragma unroll
    for (int t = 0; t < 2; ++t) {
      v8f acc = {};
      const int jl = t * 16 + col;
      #pragma unroll
      for (int os = 0; os < 16; ++os) {
        Frag16 kf;
        kf.q[0] = *(const v4u*)&ldsK[cur][jl][os * 32 + g * 8];
        kf.q[1] = *(const v4u*)&ldsK[cur][jl][os * 32 + 16 + g * 8];
        acc = wmma_f16(qf[os].v, kf.v, acc);
      }
      s[t] = acc;
    }

    // Online softmax: rows r+g*8 live in lane groups of 16 -> xor reduce
    float scale[8];
    #pragma unroll
    for (int r = 0; r < 8; ++r) {
      float rm = fmaxf(s[0][r], s[1][r]);
      rm = fmaxf(rm, __shfl_xor(rm, 1, 16));
      rm = fmaxf(rm, __shfl_xor(rm, 2, 16));
      rm = fmaxf(rm, __shfl_xor(rm, 4, 16));
      rm = fmaxf(rm, __shfl_xor(rm, 8, 16));
      float mnew = fmaxf(mrow[r], rm);
      scale[r] = __expf(mrow[r] - mnew);
      mrow[r] = mnew;
      float p0 = __expf(s[0][r] - mnew);
      float p1 = __expf(s[1][r] - mnew);
      s[0][r] = p0; s[1][r] = p1;
      float rs = p0 + p1;
      rs += __shfl_xor(rs, 1, 16);
      rs += __shfl_xor(rs, 2, 16);
      rs += __shfl_xor(rs, 4, 16);
      rs += __shfl_xor(rs, 8, 16);
      lrow[r] = lrow[r] * scale[r] + rs;
    }
    #pragma unroll
    for (int ct = 0; ct < 16; ++ct) {
      #pragma unroll
      for (int r = 0; r < 8; ++r)
        Oacc[ct][r] *= scale[r];
    }

    // Re-layout P (D-layout -> A-fragment) via per-wave LDS scratch
    #pragma unroll
    for (int t = 0; t < 2; ++t) {
      #pragma unroll
      for (int r = 0; r < 8; ++r)
        ldsP[wave][r + g * 8][t * 16 + col] = (_Float16)s[t][r];
    }
    asm volatile("s_wait_dscnt 0x0" ::: "memory");
    Frag16 pf;
    pf.q[0] = *(const v4u*)&ldsP[wave][col][g * 8];
    pf.q[1] = *(const v4u*)&ldsP[wave][col][16 + g * 8];

    // O += P (16x32) * V^T (32x256)
    #pragma unroll
    for (int ct = 0; ct < 16; ++ct) {
      Frag16 vf;
      vf.q[0] = *(const v4u*)&ldsV[cur][ct * 16 + col][g * 8];
      vf.q[1] = *(const v4u*)&ldsV[cur][ct * 16 + col][16 + g * 8];
      Oacc[ct] = wmma_f16(pf.v, vf.v, Oacc[ct]);
    }
    cur ^= 1;
  }

  // Epilogue: out[b][c][n] = O/l + x   (lane: col c, rows rowBase + r + g*8)
  {
    const size_t base = (size_t)b * Cc * Nn;
    #pragma unroll
    for (int ct = 0; ct < 16; ++ct) {
      const int c = ct * 16 + col;
      const size_t off = base + (size_t)c * Nn + rowBase + g * 8;
      #pragma unroll
      for (int r = 0; r < 8; ++r)
        outp[off + r] = Oacc[ct][r] / lrow[r] + x[off + r];
    }
  }
}

// ---------------------------------------------------------------------------
extern "C" void kernel_launch(void* const* d_in, const int* in_sizes, int n_in,
                              void* d_out, int out_size, void* d_ws, size_t ws_size,
                              hipStream_t stream) {
  const float* x  = (const float*)d_in[0];
  const float* Wq = (const float*)d_in[1];
  const float* bq = (const float*)d_in[2];
  const float* Wk = (const float*)d_in[3];
  const float* bk = (const float*)d_in[4];
  const float* Wv = (const float*)d_in[5];
  const float* bv = (const float*)d_in[6];
  float* out = (float*)d_out;

  // Workspace layout (f16): Q (B,N,512) | Kt (B,N,512) | V (B,256,N) ~ 80 MB
  _Float16* Qh = (_Float16*)d_ws;
  _Float16* Kh = Qh + (size_t)Bn * Nn * O2;
  _Float16* Vh = Kh + (size_t)Bn * Nn * O2;

  dim3 blk(256);
  proj_kernel<<<dim3(Nn / 16, O2 / 128, Bn), blk, 0, stream>>>(x, Wq, bq, Qh, O2, 0);
  proj_kernel<<<dim3(Nn / 16, O2 / 128, Bn), blk, 0, stream>>>(x, Wk, bk, Kh, O2, 0);
  proj_kernel<<<dim3(Nn / 16, Cc / 128, Bn), blk, 0, stream>>>(x, Wv, bv, Vh, Cc, 1);
  attn_kernel<<<dim3(Nn / 128, Bn), blk, 0, stream>>>(Qh, Kh, Vh, x, out);
}
